// GIN_45689862095186
// MI455X (gfx1250) — matrix operand order repaired and checked by reading, not compiled, not run on hardware
//
#include <hip/hip_runtime.h>
#include <hip/hip_bf16.h>

// ---------------------------------------------------------------------------
// GIN forward for MI455X (gfx1250, wave32).
// GEMMs run on v_wmma_f32_16x16x32_bf16 (bf16 inputs, f32 accumulate).
// Scatter/segment ops use global f32 atomics (memory-bound part of the graph).
// ---------------------------------------------------------------------------

#define GIN_N   100000
#define GIN_E   800000
#define GIN_FIN 64
#define GIN_DIM 256
#define GIN_C   10
#define GIN_G   64
#define BN_EPS  1e-5f

typedef __attribute__((ext_vector_type(16))) __bf16 v16bf;
typedef __attribute__((ext_vector_type(8)))  float  v8f;

// ---------------------------------------------------------------------------
// Pack a row-major f32 weight matrix W[K][256] into WMMA B-fragment order:
// chunk(nt, kt) = 32 lanes x 16 bf16, contiguous.  This code's own k order (contiguous, not the ISA table), for
// the 16-bit 32x16 B matrix: lanes 0-15 hold K=0..15 (n = lane), lanes 16-31
// hold K=16..31; element j within the 16-value lane chunk is k_local = j.
// ---------------------------------------------------------------------------
__global__ void k_pack_w(const float* __restrict__ W, __bf16* __restrict__ out,
                         int K, int Ncols) {
  int nkt = K >> 5;                      // K tiles of 32
  int total = (Ncols >> 4) * nkt * 32;   // one thread per (nt, kt, lane)
  int idx = blockIdx.x * blockDim.x + threadIdx.x;
  if (idx >= total) return;
  int lane = idx & 31;
  int kt   = (idx >> 5) % nkt;
  int nt   = (idx >> 5) / nkt;
  int n    = nt * 16 + (lane & 15);
  int kb   = kt * 32 + ((lane >> 4) << 4);
  __bf16* o = out + (size_t)idx * 16;
#pragma unroll
  for (int j = 0; j < 16; ++j)
    o[j] = (__bf16)W[(size_t)(kb + j) * Ncols + n];
}

// ---------------------------------------------------------------------------
// WMMA GEMM:  Out[M x 256] = relu?( A[M x K] * B + bias )
// Block = 256 threads = 8 waves; each wave computes a 16 x 256 slab
// (16 accumulators of v8f).  K in steps of 32 via wmma_f32_16x16x32_bf16.
// ---------------------------------------------------------------------------
__global__ void __launch_bounds__(256)
k_gemm_bias_act(const float* __restrict__ A, const __bf16* __restrict__ Bp,
                const float* __restrict__ bias, float* __restrict__ Out,
                int M, int K, int relu) {
  const int lane   = threadIdx.x & 31;
  const int waveId = threadIdx.x >> 5;
  const int rowBase = blockIdx.x * 128 + waveId * 16;
  const int m    = lane & 15;
  const int half = lane >> 4;
  int row  = rowBase + m;
  int rowc = row < M ? row : (M - 1);     // clamp OOB loads; stores guarded
  const int nkt = K >> 5;

  v8f acc[16];
#pragma unroll
  for (int i = 0; i < 16; ++i) acc[i] = (v8f){0.f,0.f,0.f,0.f,0.f,0.f,0.f,0.f};

  for (int kt = 0; kt < nkt; ++kt) {
    // A fragment (16x32 bf16, contiguous layout): VGPR i holds K pair
    // k = half*16 + i*2  (+ kt*32)
    v16bf afrag;
    const float* arow = A + (size_t)rowc * K + kt * 32 + half * 16;
#pragma unroll
    for (int i = 0; i < 8; ++i) {
      float2 f = *(const float2*)(arow + (i << 1));
      afrag[2 * i]     = (__bf16)f.x;
      afrag[2 * i + 1] = (__bf16)f.y;
    }
    const __bf16* bbase = Bp + ((size_t)kt * 32 + lane) * 16;
#pragma unroll
    for (int nt = 0; nt < 16; ++nt) {
      v16bf bfrag = *(const v16bf*)(bbase + (size_t)nt * nkt * 32 * 16);
      acc[nt] = __builtin_amdgcn_wmma_f32_16x16x32_bf16(
          false, afrag, false, bfrag, (short)0, acc[nt], false, false);
    }
  }

  if (rowBase < M) {
#pragma unroll
    for (int nt = 0; nt < 16; ++nt) {
      int n = nt * 16 + (lane & 15);
      float bv = bias[n];
#pragma unroll
      for (int i = 0; i < 8; ++i) {        // C/D layout: M = i + half*8
        int r = rowBase + half * 8 + i;
        if (r < M) {
          float v = acc[nt][i] + bv;
          if (relu) v = fmaxf(v, 0.f);
          Out[(size_t)r * GIN_DIM + n] = v;
        }
      }
    }
  }
}

// ---------------------------------------------------------------------------
// Elementwise / graph kernels
// ---------------------------------------------------------------------------
__global__ void k_zero(float* p, int n) {
  int i = blockIdx.x * blockDim.x + threadIdx.x;
  if (i < n) p[i] = 0.f;
}

__global__ void k_copy4(const float* __restrict__ in, float* __restrict__ out,
                        long long n4) {
  long long i = (long long)blockIdx.x * blockDim.x + threadIdx.x;
  if (i < n4) ((float4*)out)[i] = ((const float4*)in)[i];
}

// agg[dst] += h[src]; one thread per (edge, 4-feature chunk)
__global__ void k_scatter_add(const float* __restrict__ H,
                              float* __restrict__ Agg,
                              const int* __restrict__ src,
                              const int* __restrict__ dst,
                              int dim, int chunkShift) {
  long long idx = (long long)blockIdx.x * blockDim.x + threadIdx.x;
  long long total = (long long)GIN_E << chunkShift;
  if (idx >= total) return;
  int e = (int)(idx >> chunkShift);
  int c = ((int)idx & ((1 << chunkShift) - 1)) << 2;
  int s = src[e], d = dst[e];
  const float4 v = *(const float4*)(H + (size_t)s * dim + c);
  float* o = Agg + (size_t)d * dim + c;
  atomicAdd(o + 0, v.x);
  atomicAdd(o + 1, v.y);
  atomicAdd(o + 2, v.z);
  atomicAdd(o + 3, v.w);
}

// per-channel sum / sumsq over N rows (thread t <-> channel t)
__global__ void __launch_bounds__(256)
k_bn_stats(const float* __restrict__ H, float* __restrict__ stats, int M) {
  int c = threadIdx.x;
  int rowsPer = (M + gridDim.x - 1) / gridDim.x;
  int r0 = blockIdx.x * rowsPer;
  int r1 = r0 + rowsPer; if (r1 > M) r1 = M;
  float s = 0.f, ss = 0.f;
  for (int r = r0; r < r1; ++r) {
    float v = H[(size_t)r * GIN_DIM + c];
    s += v; ss += v * v;
  }
  atomicAdd(&stats[c], s);
  atomicAdd(&stats[GIN_DIM + c], ss);
}

__global__ void k_bn_apply(float* __restrict__ H, const float* __restrict__ stats,
                           const float* __restrict__ gamma,
                           const float* __restrict__ beta, int M) {
  long long i = (long long)blockIdx.x * blockDim.x + threadIdx.x;
  long long total = (long long)M * GIN_DIM;
  if (i >= total) return;
  int c = (int)(i & (GIN_DIM - 1));
  float mean = stats[c] * (1.f / (float)M);
  float var  = stats[GIN_DIM + c] * (1.f / (float)M) - mean * mean;
  float sc   = rsqrtf(var + BN_EPS) * gamma[c];
  H[i] = (H[i] - mean) * sc + beta[c];
}

// segment-sum pool over sorted batch ids; one thread per (node, 4-chan chunk)
__global__ void k_pool_scatter(const float* __restrict__ H,
                               const int* __restrict__ batch,
                               float* __restrict__ psum,
                               float* __restrict__ pcnt) {
  long long i = (long long)blockIdx.x * blockDim.x + threadIdx.x;
  long long total = (long long)GIN_N * (GIN_DIM / 4);
  if (i >= total) return;
  int node = (int)(i >> 6);
  int c = ((int)i & 63) << 2;
  int b = batch[node];
  const float4 v = *(const float4*)(H + (size_t)node * GIN_DIM + c);
  float* o = psum + (size_t)b * GIN_DIM + c;
  atomicAdd(o + 0, v.x);
  atomicAdd(o + 1, v.y);
  atomicAdd(o + 2, v.z);
  atomicAdd(o + 3, v.w);
  if (c == 0) atomicAdd(&pcnt[b], 1.f);
}

__global__ void k_pool_div(const float* __restrict__ psum,
                           const float* __restrict__ pcnt,
                           float* __restrict__ pooled) {
  int i = blockIdx.x * blockDim.x + threadIdx.x;
  if (i >= GIN_G * GIN_DIM) return;
  int g = i >> 8;
  float cnt = pcnt[g]; if (cnt < 1.f) cnt = 1.f;
  pooled[i] = psum[i] / cnt;
}

// fc2 (256 -> 10) + log_softmax; one thread per graph row (64 rows total)
__global__ void k_fc2_logsoftmax(const float* __restrict__ Hin,
                                 const float* __restrict__ W2,
                                 const float* __restrict__ b2,
                                 float* __restrict__ out) {
  int r = blockIdx.x * blockDim.x + threadIdx.x;
  if (r >= GIN_G) return;
  float logits[GIN_C];
#pragma unroll
  for (int c = 0; c < GIN_C; ++c) {
    float a = b2[c];
    for (int k = 0; k < GIN_DIM; ++k)
      a += Hin[(size_t)r * GIN_DIM + k] * W2[(size_t)k * GIN_C + c];
    logits[c] = a;
  }
  float mx = logits[0];
#pragma unroll
  for (int c = 1; c < GIN_C; ++c) mx = fmaxf(mx, logits[c]);
  float se = 0.f;
#pragma unroll
  for (int c = 0; c < GIN_C; ++c) se += __expf(logits[c] - mx);
  float lse = mx + __logf(se);
#pragma unroll
  for (int c = 0; c < GIN_C; ++c) out[(size_t)r * GIN_C + c] = logits[c] - lse;
}

// ---------------------------------------------------------------------------
// Host-side orchestration
// ---------------------------------------------------------------------------
static inline int cdiv(long long a, long long b) { return (int)((a + b - 1) / b); }

extern "C" void kernel_launch(void* const* d_in, const int* in_sizes, int n_in,
                              void* d_out, int out_size, void* d_ws, size_t ws_size,
                              hipStream_t stream) {
  (void)in_sizes; (void)n_in; (void)out_size; (void)ws_size;

  const float* x     = (const float*)d_in[0];
  const int*   ei    = (const int*)d_in[1];
  const int*   src   = ei;             // edge_index[0]
  const int*   dst   = ei + GIN_E;     // edge_index[1]
  const int*   batch = (const int*)d_in[2];

  struct Layer { const float *Wa, *ba, *Wb, *bb, *gamma, *beta; } L[5];
  for (int l = 0; l < 5; ++l) {
    int b = 3 + l * 6;
    L[l].Wa = (const float*)d_in[b + 0];
    L[l].ba = (const float*)d_in[b + 1];
    L[l].Wb = (const float*)d_in[b + 2];
    L[l].bb = (const float*)d_in[b + 3];
    L[l].gamma = (const float*)d_in[b + 4];
    L[l].beta  = (const float*)d_in[b + 5];
  }
  const float* fc1W = (const float*)d_in[33];
  const float* fc1b = (const float*)d_in[34];
  const float* fc2W = (const float*)d_in[35];
  const float* fc2b = (const float*)d_in[36];

  // workspace layout
  char* ws = (char*)d_ws;
  size_t off = 0;
  auto carve = [&](size_t bytes) -> char* {
    char* p = ws + off;
    off += (bytes + 255) & ~(size_t)255;
    return p;
  };
  float* agg  = (float*)carve((size_t)GIN_N * GIN_DIM * 4);
  float* tbuf = (float*)carve((size_t)GIN_N * GIN_DIM * 4);
  float* hbuf = (float*)carve((size_t)GIN_N * GIN_DIM * 4);
  float* stats  = (float*)carve(2 * GIN_DIM * 4);
  float* psum   = (float*)carve((size_t)GIN_G * GIN_DIM * 4);
  float* pcnt   = (float*)carve(GIN_G * 4);
  float* pooled = (float*)carve((size_t)GIN_G * GIN_DIM * 4);
  float* fcbuf  = (float*)carve((size_t)GIN_G * GIN_DIM * 4);
  __bf16* WaP[5]; __bf16* WbP[5]; __bf16* fc1P;
  WaP[0] = (__bf16*)carve((size_t)GIN_FIN * GIN_DIM * 2);
  for (int l = 1; l < 5; ++l) WaP[l] = (__bf16*)carve((size_t)GIN_DIM * GIN_DIM * 2);
  for (int l = 0; l < 5; ++l) WbP[l] = (__bf16*)carve((size_t)GIN_DIM * GIN_DIM * 2);
  fc1P = (__bf16*)carve((size_t)GIN_DIM * GIN_DIM * 2);

  // --- pack weights to bf16 B-fragment layout ---
  {
    int t64  = (GIN_DIM / 16) * (GIN_FIN / 32) * 32;   // K=64 pack threads
    int t256 = (GIN_DIM / 16) * (GIN_DIM / 32) * 32;   // K=256 pack threads
    k_pack_w<<<cdiv(t64, 256),  256, 0, stream>>>(L[0].Wa, WaP[0], GIN_FIN, GIN_DIM);
    for (int l = 1; l < 5; ++l)
      k_pack_w<<<cdiv(t256, 256), 256, 0, stream>>>(L[l].Wa, WaP[l], GIN_DIM, GIN_DIM);
    for (int l = 0; l < 5; ++l)
      k_pack_w<<<cdiv(t256, 256), 256, 0, stream>>>(L[l].Wb, WbP[l], GIN_DIM, GIN_DIM);
    k_pack_w<<<cdiv(t256, 256), 256, 0, stream>>>(fc1W, fc1P, GIN_DIM, GIN_DIM);
  }

  const int gemmGrid = cdiv(GIN_N, 128);

  for (int l = 0; l < 5; ++l) {
    const float* hin = (l == 0) ? x : hbuf;
    int dim = (l == 0) ? GIN_FIN : GIN_DIM;
    int chunkShift = (dim == 64) ? 4 : 6;     // dim/4 chunks per edge

    // agg = h  (GIN eps=0: x + sum of neighbors)
    long long n4 = (long long)GIN_N * dim / 4;
    k_copy4<<<cdiv(n4, 256), 256, 0, stream>>>(hin, agg, n4);

    // agg[dst] += h[src]
    long long st = (long long)GIN_E << chunkShift;
    k_scatter_add<<<cdiv(st, 256), 256, 0, stream>>>(hin, agg, src, dst, dim, chunkShift);

    // t = relu(agg @ Wa + ba) ; h = relu(t @ Wb + bb)  (outer F.relu fused)
    k_gemm_bias_act<<<gemmGrid, 256, 0, stream>>>(agg, WaP[l], L[l].ba, tbuf,
                                                  GIN_N, dim, 1);
    k_gemm_bias_act<<<gemmGrid, 256, 0, stream>>>(tbuf, WbP[l], L[l].bb, hbuf,
                                                  GIN_N, GIN_DIM, 1);

    // BatchNorm (training-mode batch stats, biased variance)
    k_zero<<<cdiv(2 * GIN_DIM, 256), 256, 0, stream>>>(stats, 2 * GIN_DIM);
    k_bn_stats<<<240, 256, 0, stream>>>(hbuf, stats, GIN_N);
    long long tot = (long long)GIN_N * GIN_DIM;
    k_bn_apply<<<cdiv(tot, 256), 256, 0, stream>>>(hbuf, stats, L[l].gamma,
                                                   L[l].beta, GIN_N);
  }

  // global mean pool
  k_zero<<<cdiv(GIN_G * GIN_DIM + GIN_G, 256), 256, 0, stream>>>(psum, GIN_G * GIN_DIM + GIN_G);
  long long pt = (long long)GIN_N * (GIN_DIM / 4);
  k_pool_scatter<<<cdiv(pt, 256), 256, 0, stream>>>(hbuf, batch, psum, pcnt);
  k_pool_div<<<cdiv(GIN_G * GIN_DIM, 256), 256, 0, stream>>>(psum, pcnt, pooled);

  // fc1 (WMMA GEMM, M=64) + relu
  k_gemm_bias_act<<<1, 256, 0, stream>>>(pooled, fc1P, fc1b, fcbuf, GIN_G, GIN_DIM, 1);

  // fc2 + log_softmax
  k_fc2_logsoftmax<<<1, 64, 0, stream>>>(fcbuf, fc2W, fc2b, (float*)d_out);
}
